// UniversalLinearEmbeddingAE_12378095747124
// MI455X (gfx1250) — matrix operand order repaired and checked by reading, not compiled, not run on hardware
//
#include <hip/hip_runtime.h>

typedef __attribute__((ext_vector_type(16))) _Float16 v16h;
typedef __attribute__((ext_vector_type(8)))  _Float16 v8h;
typedef __attribute__((ext_vector_type(8)))  float    v8f;

// ---------------------------------------------------------------------------
// Weight convert: f32 [din,dout] row-major  ->  f16 transposed padded
// wt[n*din_p + k] = w[k*dout + n]  (zero outside valid range)
// ---------------------------------------------------------------------------
__global__ void ulae_convert_weight(const float* __restrict__ w,
                                    _Float16* __restrict__ wt,
                                    int din, int dout, int din_p, int dout_p) {
    int i = blockIdx.x * blockDim.x + threadIdx.x;
    int total = din_p * dout_p;
    if (i >= total) return;
    int k = i % din_p;
    int n = i / din_p;
    float v = (k < din && n < dout) ? w[(size_t)k * dout + n] : 0.0f;
    wt[i] = (_Float16)v;
}

// f32 -> f16 straight copy (input x, width 64 already a multiple of 32)
__global__ void ulae_convert_in(const float* __restrict__ x,
                                _Float16* __restrict__ a, long n) {
    long i = (long)blockIdx.x * blockDim.x + threadIdx.x;
    if (i >= n) return;
    a[i] = (_Float16)x[i];
}

// ---------------------------------------------------------------------------
// WMMA GEMM:  Out[rows, dout_p] = A[rows, din_p] * W[din_p, dout_p]
// A: f16 row-major (K padded with zeros).  Wt: f16 TRANSPOSED [dout_p, din_p].
// Each wave: MT=2 16-row tiles x NT 16-col tiles, K loop step 32.
// B fragments reused across both row tiles: 1.5 b128 loads per WMMA at NT=4.
// MODE 0: relu -> f16,  MODE 1: f16,  MODE 2: f32 (final output)
// ---------------------------------------------------------------------------
template <int NT, int MODE>
__global__ __launch_bounds__(256) void ulae_gemm_wmma(
    const _Float16* __restrict__ A, int din_p,
    const _Float16* __restrict__ Wt,
    void* __restrict__ Out, int ldo,
    int rows, int dout_p) {
    const int wave = threadIdx.x >> 5;
    const int lane = threadIdx.x & 31;
    const int mgroups = rows >> 5;            // rows is a multiple of 32
    const int ngroups = dout_p / (16 * NT);
    const long tile = (long)blockIdx.x * (blockDim.x >> 5) + wave;
    if (tile >= (long)mgroups * ngroups) return;   // wave-uniform
    const int mt = (int)(tile % mgroups) * 2;      // first of two m-tiles
    const int ng = (int)(tile / mgroups);
    const int n0 = ng * 16 * NT;
    const bool hi = lane >= 16;
    const int  lr = lane & 15;

    v8f acc0[NT], acc1[NT];
#pragma unroll
    for (int i = 0; i < NT; i++) { acc0[i] = 0.0f; acc1[i] = 0.0f; }

    // A fragment: lanes 0-15 row M=lr hold K 0..15; lanes 16-31 hold K 16..31
    const _Float16* arow0 = A + (size_t)(mt * 16 + lr) * din_p + (hi ? 16 : 0);
    const _Float16* arow1 = arow0 + (size_t)16 * din_p;
    // B fragment: lanes 0-15 col N hold K 0..15; lanes 16-31 hold K 16..31
    const _Float16* bbase = Wt + (size_t)(n0 + lr) * din_p + (hi ? 16 : 0);

    for (int k0 = 0; k0 < din_p; k0 += 32) {
        v8h c10 = *(const v8h*)(arow0 + k0);
        v8h c20 = *(const v8h*)(arow0 + k0 + 8);
        v8h c11 = *(const v8h*)(arow1 + k0);
        v8h c21 = *(const v8h*)(arow1 + k0 + 8);
        v16h a0, a1;
#pragma unroll
        for (int i = 0; i < 8; i++) {
            a0[i] = c10[i]; a0[i + 8] = c20[i];
            a1[i] = c11[i]; a1[i + 8] = c21[i];
        }
#pragma unroll
        for (int nt = 0; nt < NT; nt++) {
            v16h b = *(const v16h*)(bbase + (size_t)nt * 16 * din_p + k0);
            acc0[nt] = __builtin_amdgcn_wmma_f32_16x16x32_f16(
                false, a0, false, b, (short)0, acc0[nt], false, false);
            acc1[nt] = __builtin_amdgcn_wmma_f32_16x16x32_f16(
                false, a1, false, b, (short)0, acc1[nt], false, false);
        }
    }

    // C/D layout: VGPR r -> (M = r + (hi?8:0), N = lr)
#pragma unroll
    for (int nt = 0; nt < NT; nt++) {
        int col = n0 + nt * 16 + lr;
#pragma unroll
        for (int r = 0; r < 8; r++) {
            int row0 = mt * 16 + r + (hi ? 8 : 0);
            int row1 = row0 + 16;
            float v0 = acc0[nt][r];
            float v1 = acc1[nt][r];
            if (MODE == 0) {
                v0 = v0 > 0.0f ? v0 : 0.0f;
                v1 = v1 > 0.0f ? v1 : 0.0f;
            }
            if (MODE == 2) {
                ((float*)Out)[(size_t)row0 * ldo + col] = v0;
                ((float*)Out)[(size_t)row1 * ldo + col] = v1;
            } else {
                ((_Float16*)Out)[(size_t)row0 * ldo + col] = (_Float16)v0;
                ((_Float16*)Out)[(size_t)row1 * ldo + col] = (_Float16)v1;
            }
        }
    }
}

// ---------------------------------------------------------------------------
// Koopman step:  next = exp(mu) * R(omega) * emb,  scattered to time-concat
// ---------------------------------------------------------------------------
__global__ void ulae_koopman(const _Float16* __restrict__ emb,
                             const _Float16* __restrict__ lam,
                             _Float16* __restrict__ nxt,
                             int s, int NE, int T, int S) {
    int r = blockIdx.x * blockDim.x + threadIdx.x;
    if (r >= NE) return;
    float e0 = (float)emb[(size_t)r * 32 + 0];
    float e1 = (float)emb[(size_t)r * 32 + 1];
    float mu = (float)lam[(size_t)r * 32 + 0];
    float om = (float)lam[(size_t)r * 32 + 1];
    float sc = __expf(mu);
    float c  = __cosf(om);
    float sn = __sinf(om);
    float nx = sc * (c * e0 - sn * e1);
    float ny = sc * (sn * e0 + c * e1);
    int bk = r / T, t = r % T;
    size_t r2 = ((size_t)bk * S + s) * T + t;   // concat along time
    _Float16* p = nxt + r2 * 32;
    p[0] = (_Float16)nx;
    p[1] = (_Float16)ny;
#pragma unroll
    for (int j = 2; j < 32; j++) p[j] = (_Float16)0.0f;  // keep K-pad zero
}

// ---------------------------------------------------------------------------
extern "C" void kernel_launch(void* const* d_in, const int* in_sizes, int n_in,
                              void* d_out, int out_size, void* d_ws, size_t ws_size,
                              hipStream_t stream) {
    (void)in_sizes; (void)n_in; (void)out_size;
    const int NE = 32 * 4 * 512;   // 65536 rows through encoder/aux
    const int T = 512, S = 4;
    const int ND = NE * S;         // 262144 rows through decoder

    static const int ENCD[9] = {64, 1024, 512, 256, 128, 64, 32, 16, 2};
    static const int AUXD[5] = {2, 64, 128, 256, 2};
    static const int DECD[9] = {2, 8, 16, 32, 64, 128, 256, 512, 64};
    auto pad32 = [](int v) { return (v + 31) & ~31; };

    // input pointer table: x, enc[8], aux[4][4], dec[8]
    const float* x = (const float*)d_in[0];
    const float* wenc[8]; const float* waux[4][4]; const float* wdec[8];
    {
        int p = 1;
        for (int i = 0; i < 8; i++) wenc[i] = (const float*)d_in[p++];
        for (int s = 0; s < 4; s++)
            for (int i = 0; i < 4; i++) waux[s][i] = (const float*)d_in[p++];
        for (int i = 0; i < 8; i++) wdec[i] = (const float*)d_in[p++];
    }

    // -------- scratch layout --------
    char* ws = (char*)d_ws;
    size_t off = 0;
    auto alloc = [&](size_t bytes) -> char* {
        char* r = ws + off;
        off = (off + bytes + 255) & ~(size_t)255;
        return r;
    };
    _Float16* wt_enc[8]; _Float16* wt_aux[4][4]; _Float16* wt_dec[8];
    for (int i = 0; i < 8; i++)
        wt_enc[i] = (_Float16*)alloc((size_t)pad32(ENCD[i]) * pad32(ENCD[i + 1]) * 2);
    for (int s = 0; s < 4; s++)
        for (int i = 0; i < 4; i++)
            wt_aux[s][i] = (_Float16*)alloc((size_t)pad32(AUXD[i]) * pad32(AUXD[i + 1]) * 2);
    for (int i = 0; i < 8; i++)
        wt_dec[i] = (_Float16*)alloc((size_t)pad32(DECD[i]) * pad32(DECD[i + 1]) * 2);
    _Float16* xh   = (_Float16*)alloc((size_t)NE * 64 * 2);
    _Float16* embb = (_Float16*)alloc((size_t)NE * 32 * 2);
    _Float16* lamb = (_Float16*)alloc((size_t)NE * 32 * 2);
    _Float16* nxt  = (_Float16*)alloc((size_t)ND * 32 * 2);
    size_t rem  = (ws_size > off) ? (ws_size - off) : 0;
    size_t bufB = (rem / 2) & ~(size_t)255;
    _Float16* ping = (_Float16*)(ws + off);
    _Float16* pong = (_Float16*)(ws + off + bufB);

    // chunks are multiples of 32 rows so MT=2 tiling is always exact
    auto chunkRows = [&](int maxW, int total) -> int {
        long r = (long)(bufB / ((size_t)maxW * 2));
        r &= ~31L;
        if (r < 32) r = 32;
        if (r > total) r = total;
        return (int)r;
    };

    // -------- GEMM dispatch --------
    auto gemm = [&](const _Float16* A, const _Float16* Wt, int din_p,
                    void* out, int ldo, int rows, int dout_p, int mode) {
        int NT = (dout_p % 64 == 0) ? 4 : 2;
        long tiles = (long)(rows / 32) * (dout_p / (16 * NT));
        dim3 grid((unsigned)((tiles + 7) / 8)), block(256);
        int key = mode * 10 + NT;
        switch (key) {
        case  4: ulae_gemm_wmma<4, 0><<<grid, block, 0, stream>>>(A, din_p, Wt, out, ldo, rows, dout_p); break;
        case  2: ulae_gemm_wmma<2, 0><<<grid, block, 0, stream>>>(A, din_p, Wt, out, ldo, rows, dout_p); break;
        case 14: ulae_gemm_wmma<4, 1><<<grid, block, 0, stream>>>(A, din_p, Wt, out, ldo, rows, dout_p); break;
        case 12: ulae_gemm_wmma<2, 1><<<grid, block, 0, stream>>>(A, din_p, Wt, out, ldo, rows, dout_p); break;
        case 24: ulae_gemm_wmma<4, 2><<<grid, block, 0, stream>>>(A, din_p, Wt, out, ldo, rows, dout_p); break;
        default: ulae_gemm_wmma<2, 2><<<grid, block, 0, stream>>>(A, din_p, Wt, out, ldo, rows, dout_p); break;
        }
    };

    // -------- weight + input conversion --------
    {
        long n = (long)NE * 64;
        ulae_convert_in<<<(unsigned)((n + 255) / 256), 256, 0, stream>>>(x, xh, n);
        for (int i = 0; i < 8; i++) {
            int dp = pad32(ENCD[i]), op = pad32(ENCD[i + 1]);
            ulae_convert_weight<<<(dp * op + 255) / 256, 256, 0, stream>>>(
                wenc[i], wt_enc[i], ENCD[i], ENCD[i + 1], dp, op);
        }
        for (int s = 0; s < 4; s++)
            for (int i = 0; i < 4; i++) {
                int dp = pad32(AUXD[i]), op = pad32(AUXD[i + 1]);
                ulae_convert_weight<<<(dp * op + 255) / 256, 256, 0, stream>>>(
                    waux[s][i], wt_aux[s][i], AUXD[i], AUXD[i + 1], dp, op);
            }
        for (int i = 0; i < 8; i++) {
            int dp = pad32(DECD[i]), op = pad32(DECD[i + 1]);
            ulae_convert_weight<<<(dp * op + 255) / 256, 256, 0, stream>>>(
                wdec[i], wt_dec[i], DECD[i], DECD[i + 1], dp, op);
        }
    }

    // -------- encoder: 64 ->...-> 2 (padded 32), emb to embb --------
    {
        int cr = chunkRows(1024, NE);
        for (int cs = 0; cs < NE; cs += cr) {
            int rows = (NE - cs < cr) ? (NE - cs) : cr;
            const _Float16* in = xh + (size_t)cs * 64;
            int din_p = 64;
            _Float16* bufs[2] = {ping, pong};
            int pi = 0;
            for (int i = 0; i < 8; i++) {
                int dout_p = pad32(ENCD[i + 1]);
                bool last = (i == 7);
                void* out = last ? (void*)(embb + (size_t)cs * 32) : (void*)bufs[pi];
                gemm(in, wt_enc[i], din_p, out, dout_p, rows, dout_p, last ? 1 : 0);
                in = bufs[pi];
                pi ^= 1;
                din_p = dout_p;
            }
        }
    }

    // -------- aux nets + Koopman rotation, scatter into nxt --------
    {
        int cr = chunkRows(256, NE);
        for (int s = 0; s < 4; s++) {
            for (int cs = 0; cs < NE; cs += cr) {
                int rows = (NE - cs < cr) ? (NE - cs) : cr;
                const _Float16* in = embb + (size_t)cs * 32;
                int din_p = 32;
                _Float16* bufs[2] = {ping, pong};
                int pi = 0;
                for (int i = 0; i < 4; i++) {
                    int dout_p = pad32(AUXD[i + 1]);
                    bool last = (i == 3);
                    void* out = last ? (void*)(lamb + (size_t)cs * 32) : (void*)bufs[pi];
                    gemm(in, wt_aux[s][i], din_p, out, dout_p, rows, dout_p, last ? 1 : 0);
                    in = bufs[pi];
                    pi ^= 1;
                    din_p = dout_p;
                }
            }
            ulae_koopman<<<(NE + 255) / 256, 256, 0, stream>>>(embb, lamb, nxt, s, NE, T, S);
        }
    }

    // -------- decoder: 2 (padded 32) ->...-> 64, f32 out --------
    {
        int cr = chunkRows(512, ND);
        for (int cs = 0; cs < ND; cs += cr) {
            int rows = (ND - cs < cr) ? (ND - cs) : cr;
            const _Float16* in = nxt + (size_t)cs * 32;
            int din_p = 32;
            _Float16* bufs[2] = {ping, pong};
            int pi = 0;
            for (int i = 0; i < 8; i++) {
                int dout_p = pad32(DECD[i + 1]);
                bool last = (i == 7);
                void* out = last ? (void*)((float*)d_out + (size_t)cs * 64) : (void*)bufs[pi];
                int ldo = last ? 64 : dout_p;
                gemm(in, wt_dec[i], din_p, out, ldo, rows, dout_p, last ? 2 : 0);
                in = bufs[pi];
                pi ^= 1;
                din_p = dout_p;
            }
        }
    }
}
